// EdgeConv_10299331576139
// MI455X (gfx1250) — matrix operand order repaired and checked by reading, not compiled, not run on hardware
//
#include <hip/hip_runtime.h>

typedef float  v2f   __attribute__((ext_vector_type(2)));
typedef float  v8f   __attribute__((ext_vector_type(8)));
typedef __bf16 v16bf __attribute__((ext_vector_type(16)));

#define NBATCH 128
#define NPTS   512
#define NCH    64
#define NK     16

// ---------------------------------------------------------------------------
// WMMA helpers
// ---------------------------------------------------------------------------
__device__ inline v8f wmma_bf16(v16bf a, v16bf b, v8f c) {
    return __builtin_amdgcn_wmma_f32_16x16x32_bf16(false, a, false, b, (short)0, c, false, false);
}

// A fragment, bf16 16x32 (MxK): lane(l&15)=row M, half=l>>4.
// VGPR0-3 hold K=[8h,8h+8), VGPR4-7 hold K=[16+8h,16+8h+8)  (two 16B chunks).
__device__ inline v16bf load_a16(const __bf16* rowp, int kbase, int half) {
    union { v16bf v; struct { uint4 lo, hi; } q; } u;
    u.q.lo = *(const uint4*)(rowp + kbase + 8 * half);
    u.q.hi = *(const uint4*)(rowp + kbase + 16 + 8 * half);
    return u.v;
}

// B fragment, bf16 32x16 (KxN) with B[k][n] = W[n][k] (row-major weights):
// per-lane data = W[n][kbase+8h .. +8) then W[n][kbase+16+8h .. +8) (two 16B chunks).
__device__ inline v16bf load_b16(const __bf16* wrow, int kbase, int half) {
    union { v16bf v; struct { uint4 lo, hi; } q; } u;
    const __bf16* p = wrow + kbase + 8 * half;
    u.q.lo = *(const uint4*)(p);
    u.q.hi = *(const uint4*)(p + 16);
    return u.v;
}

__device__ inline unsigned int f32_ordkey(float f) {
    unsigned int u = __float_as_uint(f);
    return (u & 0x80000000u) ? ~u : (u | 0x80000000u);
}

// LDS byte address for async-to-LDS ops: generic LDS pointers carry the
// wave-relative LDS offset in their low 32 bits (flat aperture mapping).
__device__ inline unsigned lds_addr32(const void* p) {
    return (unsigned)(unsigned long long)p;
}

// GLOBAL_LOAD_ASYNC_TO_LDS_B128: 16B memory -> LDS, tracked by ASYNCcnt.
__device__ inline void async_copy_b128(unsigned lds_byte_addr, const void* gsrc) {
    asm volatile("global_load_async_to_lds_b128 %0, %1, off"
                 :: "v"(lds_byte_addr), "v"((unsigned long long)gsrc)
                 : "memory");
}
__device__ inline void wait_asynccnt0() {
    asm volatile("s_wait_asynccnt 0x0" ::: "memory");
}

// ---------------------------------------------------------------------------
// Kernel 0: x (f32) -> xbf (bf16), done once so the kNN gather moves bf16 rows
// ---------------------------------------------------------------------------
__global__ void __launch_bounds__(256) xbf_k(const float* __restrict__ x,
                                             __bf16* __restrict__ xbf) {
    int i = (blockIdx.x * 256 + threadIdx.x) * 8;
    const float4* s = (const float4*)(x + i);
    float4 a = s[0], b = s[1];
    xbf[i + 0] = (__bf16)a.x; xbf[i + 1] = (__bf16)a.y;
    xbf[i + 2] = (__bf16)a.z; xbf[i + 3] = (__bf16)a.w;
    xbf[i + 4] = (__bf16)b.x; xbf[i + 5] = (__bf16)b.y;
    xbf[i + 6] = (__bf16)b.z; xbf[i + 7] = (__bf16)b.w;
}

// ---------------------------------------------------------------------------
// Kernel 1: per-point squared norms
// ---------------------------------------------------------------------------
__global__ void __launch_bounds__(256) norms_k(const float* __restrict__ x,
                                               float* __restrict__ norms) {
    int pt = blockIdx.x * 256 + threadIdx.x;
    const float4* r = (const float4*)(x + (size_t)pt * NCH);
    float s = 0.f;
#pragma unroll
    for (int i = 0; i < 16; ++i) {
        float4 v = r[i];
        s += v.x * v.x + v.y * v.y + v.z * v.z + v.w * v.w;
    }
    norms[pt] = s;
}

// ---------------------------------------------------------------------------
// Kernel 2: per-batch valid point count (mask handling for pool denominator)
// ---------------------------------------------------------------------------
__global__ void __launch_bounds__(256) nvalid_k(const unsigned char* __restrict__ mask,
                                                int* __restrict__ nvalid) {
    __shared__ int cnt;
    if (threadIdx.x == 0) cnt = 0;
    __syncthreads();
    int b = blockIdx.x;
    int local = 0;
    for (int i = threadIdx.x; i < NPTS; i += 256) local += mask[b * NPTS + i] ? 1 : 0;
    atomicAdd(&cnt, local);
    __syncthreads();
    if (threadIdx.x == 0) nvalid[b] = NPTS - cnt;
}

// ---------------------------------------------------------------------------
// Kernel 3: 16-row metric strip via f32 WMMA + wave32 top-17 selection.
// metric[j] = |xj|^2 - 2*xi.xj  (ordering-equivalent to dist; sqrt dropped)
// ---------------------------------------------------------------------------
__global__ void __launch_bounds__(512) knn_k(const float* __restrict__ x,
                                             const float* __restrict__ norms,
                                             int* __restrict__ knn) {
    __shared__ float S[16 * NPTS];       // 32 KB metric strip
    __shared__ float ncol[NPTS];         // 2 KB column norms
    const int b    = blockIdx.x >> 5;
    const int i0   = (blockIdx.x & 31) << 4;
    const int t    = threadIdx.x;
    const int wave = t >> 5, lane = t & 31;
    const int m16  = lane & 15, half = lane >> 4;

    ncol[t] = norms[b * NPTS + t];
    __syncthreads();

    const float* xb   = x + (size_t)b * NPTS * NCH;
    const float* arow = xb + (size_t)(i0 + m16) * NCH;   // A row M = lane&15

    for (int tt = 0; tt < 2; ++tt) {
        int j0 = (wave * 2 + tt) * 16;
        const float* brow = xb + (size_t)(j0 + m16) * NCH;  // B col N = lane&15
        v8f c = {0.f, 0.f, 0.f, 0.f, 0.f, 0.f, 0.f, 0.f};
#pragma unroll
        for (int kb = 0; kb < NCH; kb += 4) {
            v2f a, bb;
            a.x  = arow[kb + 2 * half];  a.y  = arow[kb + 2 * half + 1];
            bb.x = brow[kb + 2 * half];  bb.y = brow[kb + 2 * half + 1];
            c = __builtin_amdgcn_wmma_f32_16x16x4_f32(false, a, false, bb, (short)0, c, false, false);
        }
#pragma unroll
        for (int e = 0; e < 8; ++e) {
            int m = e + 8 * half;                         // C/D: row = e + 8*half
            S[m * NPTS + j0 + m16] = ncol[j0 + m16] - 2.0f * c[e];
        }
    }
    __syncthreads();

    // wave w selects top-17 minima (value, then index) for row w; first is self.
    float* row = S + wave * NPTS;
    int* dst = knn + ((size_t)(b * NPTS + i0 + wave)) * NK;
    for (int it = 0; it < NK + 1; ++it) {
        unsigned long long best = ~0ull;
        for (int q = lane; q < NPTS; q += 32) {
            unsigned long long key =
                ((unsigned long long)f32_ordkey(row[q]) << 32) | (unsigned int)q;
            best = key < best ? key : best;
        }
#pragma unroll
        for (int off = 16; off > 0; off >>= 1) {
            unsigned long long o = __shfl_xor(best, off, 32);
            best = o < best ? o : best;
        }
        int jmin = (int)(best & 0xffffffffu);
        if (lane == 0) {
            if (it > 0) dst[it - 1] = jmin;
            row[jmin] = __int_as_float(0x7f800000);       // +inf: remove from pool
        }
        __syncthreads();
    }
}

// ---------------------------------------------------------------------------
// Kernel 4: async-DMA gather + 3-layer bf16 WMMA MLP + pool + f32 WMMA residual
// Layer-1 identity: W1a*xc + W1b*(xc-xn) = (W1a+W1b)*xc + (-W1b)*xn,
// so the gather is a raw bf16 row copy (GLOBAL_LOAD_ASYNC_TO_LDS_B128).
// ---------------------------------------------------------------------------
__global__ void __launch_bounds__(256) mlp_k(const float* __restrict__ x,
                                             const __bf16* __restrict__ xbf,
                                             const float* __restrict__ W1g,
                                             const float* __restrict__ W2g,
                                             const float* __restrict__ W3g,
                                             const float* __restrict__ Wres,
                                             const int* __restrict__ knn,
                                             const int* __restrict__ nvalid,
                                             float* __restrict__ out) {
    extern __shared__ char smem[];
    float*  xcf  = (float*)(smem);               // 16x64 f32 centers      4096 B
    __bf16* xcb  = (__bf16*)(smem + 4096);       // 16x64 bf16 centers     2048 B
    __bf16* Dm   = (__bf16*)(smem + 6144);       // 256x64 nei rows / H2  32768 B
    __bf16* H1   = (__bf16*)(smem + 38912);      // 256x64 layer-1 out    32768 B
    __bf16* W1c  = (__bf16*)(smem + 71680);      // 64x64 (W1a+W1b)        8192 B
    __bf16* W1n  = (__bf16*)(smem + 79872);      // 64x64 (-W1b)           8192 B
    __bf16* W2s  = (__bf16*)(smem + 88064);      // 64x64                  8192 B
    __bf16* W3s  = (__bf16*)(smem + 96256);      // 64x64                  8192 B
    int*    idxs = (int*)(smem + 104448);        // 256 neighbor ids       1024 B
    float*  pool = (float*)(smem + 105472);      // 16x64 pooled           4096 B
    float*  den  = (float*)(smem + 109568);      // 16 denominators          64 B

    const int b  = blockIdx.x >> 5;
    const int p0 = (blockIdx.x & 31) << 4;
    const int t = threadIdx.x, wave = t >> 5, lane = t & 31;
    const int n16 = lane & 15, half = lane >> 4;

    // ---- phase A: async-stage center tiles; convert weights on VALU meanwhile
    async_copy_b128(lds_addr32(xcf) + t * 16,
                    (const char*)(x + ((size_t)b * NPTS + p0) * NCH) + t * 16);
    if (t < 128)
        async_copy_b128(lds_addr32(xcb) + t * 16,
                        (const char*)(xbf + ((size_t)b * NPTS + p0) * NCH) + t * 16);

    for (int i = t; i < 64 * 64; i += 256) {
        int o = i >> 6, k = i & 63;
        float wa = W1g[o * 128 + k], wb = W1g[o * 128 + 64 + k];
        W1c[i] = (__bf16)(wa + wb);
        W1n[i] = (__bf16)(-wb);
        W2s[i] = (__bf16)W2g[i];
        W3s[i] = (__bf16)W3g[i];
    }
    idxs[t] = knn[((size_t)(b * NPTS + p0 + (t >> 4))) * NK + (t & 15)];
    __syncthreads();

    // ---- phase B: async gather of 256 neighbor bf16 rows (128 B each) ----
    {
        int j = idxs[t];
        const char* src = (const char*)(xbf + ((size_t)b * NPTS + j) * NCH);
        unsigned dst = lds_addr32(Dm) + t * 128;
#pragma unroll
        for (int i = 0; i < 8; ++i)
            async_copy_b128(dst + i * 16, src + i * 16);
    }
    if (t < 16) {
        int nv = nvalid[b], c = 0;
        for (int kk = 0; kk < NK; ++kk) c += (idxs[t * NK + kk] < nv) ? 1 : 0;
        den[t] = (float)c;
    }
    wait_asynccnt0();
    __syncthreads();

    // ---- layer 1: (W1a+W1b)*xc + (-W1b)*xnei -> H1 ----
    for (int job = wave; job < 64; job += 8) {
        int p = job >> 2, o = job & 3;
        v8f c = {0.f, 0.f, 0.f, 0.f, 0.f, 0.f, 0.f, 0.f};
        const __bf16* ac = xcb + p * 64;                    // all 16 rows share center
        const __bf16* an = Dm + (p * 16 + n16) * 64;        // per-row neighbor
        const __bf16* wc = W1c + (o * 16 + n16) * 64;
        const __bf16* wn = W1n + (o * 16 + n16) * 64;
        c = wmma_bf16(load_a16(ac, 0, half),  load_b16(wc, 0, half),  c);
        c = wmma_bf16(load_a16(ac, 32, half), load_b16(wc, 32, half), c);
        c = wmma_bf16(load_a16(an, 0, half),  load_b16(wn, 0, half),  c);
        c = wmma_bf16(load_a16(an, 32, half), load_b16(wn, 32, half), c);
#pragma unroll
        for (int e = 0; e < 8; ++e) {
            int m = e + 8 * half;
            float v = c[e]; v = v > 0.f ? v : 0.f;
            H1[(p * 16 + m) * 64 + o * 16 + n16] = (__bf16)v;
        }
    }
    __syncthreads();

    // ---- layer 2: H1 @ W2^T -> H2 (reuses Dm) ----
    for (int job = wave; job < 64; job += 8) {
        int p = job >> 2, o = job & 3;
        v8f c = {0.f, 0.f, 0.f, 0.f, 0.f, 0.f, 0.f, 0.f};
        const __bf16* ar = H1 + (p * 16 + n16) * 64;
        const __bf16* wr = W2s + (o * 16 + n16) * 64;
        c = wmma_bf16(load_a16(ar, 0, half),  load_b16(wr, 0, half),  c);
        c = wmma_bf16(load_a16(ar, 32, half), load_b16(wr, 32, half), c);
#pragma unroll
        for (int e = 0; e < 8; ++e) {
            int m = e + 8 * half;
            float v = c[e]; v = v > 0.f ? v : 0.f;
            Dm[(p * 16 + m) * 64 + o * 16 + n16] = (__bf16)v;
        }
    }
    __syncthreads();

    // ---- layer 3 + masked mean pool over the K dimension ----
    {
        int nv = nvalid[b];
        for (int job = wave; job < 64; job += 8) {
            int p = job >> 2, o = job & 3;
            v8f c = {0.f, 0.f, 0.f, 0.f, 0.f, 0.f, 0.f, 0.f};
            const __bf16* ar = Dm + (p * 16 + n16) * 64;
            const __bf16* wr = W3s + (o * 16 + n16) * 64;
            c = wmma_bf16(load_a16(ar, 0, half),  load_b16(wr, 0, half),  c);
            c = wmma_bf16(load_a16(ar, 32, half), load_b16(wr, 32, half), c);
            float s = 0.f;
#pragma unroll
            for (int e = 0; e < 8; ++e) {
                int m = e + 8 * half;                       // neighbor index kk
                float v = c[e]; v = v > 0.f ? v : 0.f;
                s += (idxs[p * NK + m] < nv) ? v : 0.f;
            }
            s += __shfl_xor(s, 16, 32);                     // fold the two row halves
            if (half == 0) pool[p * 64 + o * 16 + n16] = s / den[p];
        }
    }
    __syncthreads();

    // ---- residual projection in f32 WMMA + final ReLU store ----
    if (wave < 4) {
        int o = wave;
        v8f c = {0.f, 0.f, 0.f, 0.f, 0.f, 0.f, 0.f, 0.f};
        const float* ar = xcf + n16 * 64;                   // A row M = lane&15
        const float* wr = Wres + (size_t)(o * 16 + n16) * 64;
#pragma unroll
        for (int kb = 0; kb < NCH; kb += 4) {
            v2f a, bb;
            a.x  = ar[kb + 2 * half];  a.y  = ar[kb + 2 * half + 1];
            bb.x = wr[kb + 2 * half];  bb.y = wr[kb + 2 * half + 1];
            c = __builtin_amdgcn_wmma_f32_16x16x4_f32(false, a, false, bb, (short)0, c, false, false);
        }
#pragma unroll
        for (int e = 0; e < 8; ++e) {
            int m = e + 8 * half;
            float v = c[e] + pool[m * 64 + o * 16 + n16];
            v = v > 0.f ? v : 0.f;
            out[((size_t)(b * NPTS + p0 + m)) * NCH + o * 16 + n16] = v;
        }
    }
}

// ---------------------------------------------------------------------------
// Launcher
// ---------------------------------------------------------------------------
extern "C" void kernel_launch(void* const* d_in, const int* in_sizes, int n_in,
                              void* d_out, int out_size, void* d_ws, size_t ws_size,
                              hipStream_t stream) {
    const float*         x    = (const float*)d_in[0];
    const unsigned char* mask = (const unsigned char*)d_in[1];
    const float*         W1   = (const float*)d_in[2];
    const float*         W2   = (const float*)d_in[3];
    const float*         W3   = (const float*)d_in[4];
    const float*         Wr   = (const float*)d_in[5];
    float*               out  = (float*)d_out;

    // ws layout: norms 256 KB | knn 4 MB | nvalid 512 B | xbf 8 MB
    float*  norms  = (float*)d_ws;
    int*    knn    = (int*)((char*)d_ws + 262144);
    int*    nvalid = (int*)((char*)d_ws + 262144 + (size_t)NBATCH * NPTS * NK * 4);
    __bf16* xbf    = (__bf16*)((char*)d_ws + 4456960);

    xbf_k<<<(NBATCH * NPTS * NCH) / (256 * 8), 256, 0, stream>>>(x, xbf);
    norms_k<<<NBATCH * NPTS / 256, 256, 0, stream>>>(x, norms);
    nvalid_k<<<NBATCH, 256, 0, stream>>>(mask, nvalid);
    knn_k<<<NBATCH * (NPTS / 16), 512, 0, stream>>>(x, norms, knn);
    mlp_k<<<NBATCH * (NPTS / 16), 256, 109632, stream>>>(x, xbf, W1, W2, W3, Wr,
                                                         knn, nvalid, out);
}
